// BlockSparseGRUv1_38010460570327
// MI455X (gfx1250) — matrix operand order repaired and checked by reading, not compiled, not run on hardware
//
#include <hip/hip_runtime.h>
#include <hip/hip_bf16.h>

// ---------------------------------------------------------------------------
// Block-sparse GRU (2 layers) for MI455X / gfx1250 (wave32, WMMA).
// Phase 1: input projections gx = X*W_ih^T + b_ih via v_wmma_f32_16x16x32_bf16
//          (16x64 macro-tile per wave: 4 WMMA per A-fragment load).
// Phase 2: persistent-scan kernel, 32 workgroups, W_hh slice (192KB bf16)
//          resident in LDS (staged with GLOBAL_LOAD_ASYNC_TO_LDS_B128);
//          per-step recurrent GEMM via WMMA; device-scope barrier per step.
// ---------------------------------------------------------------------------

typedef __attribute__((ext_vector_type(16))) __bf16 v16bf;
typedef __attribute__((ext_vector_type(8)))  __bf16 v8bf;
typedef __attribute__((ext_vector_type(8)))  float  v8f;

#define B_   16
#define T_   512
#define N_   256
#define H_   1024
#define M_   8
#define R_   (B_ * T_)   // 8192 rows (b*T + t)
#define G3_  (3 * H_)    // 3072
#define GSC  32          // persistent workgroups in scan
#define HS_  (H_ / GSC)  // 32 h-columns per workgroup
#define LB_  (H_ / M_)   // 128 block size
#define BETA_ 10.0f

typedef __attribute__((address_space(3))) unsigned char lds_byte_t;
__device__ __forceinline__ unsigned lds_off_(void* p) {
  return (unsigned)(unsigned long long)(lds_byte_t*)p;
}

__device__ __forceinline__ unsigned short f2bf(float f) {
  unsigned u = __float_as_uint(f);
  u += 0x7FFFu + ((u >> 16) & 1u);           // round-to-nearest-even
  return (unsigned short)(u >> 16);
}
__device__ __forceinline__ float sigmoidf_(float x) {
  return 1.0f / (1.0f + __expf(-x));
}

// ---------------------------------------------------------------------------
__global__ void bsgru_zero_u32(unsigned* p, int n) {
  for (int i = blockIdx.x * blockDim.x + threadIdx.x; i < n;
       i += gridDim.x * blockDim.x)
    p[i] = 0u;
}

__global__ void bsgru_cvt_bf16(const float* __restrict__ in,
                               unsigned short* __restrict__ out, int n) {
  for (int i = blockIdx.x * blockDim.x + threadIdx.x; i < n;
       i += gridDim.x * blockDim.x)
    out[i] = f2bf(in[i]);
}

// ---------------------------------------------------------------------------
// C[R x Cc] = A[R x K](bf16,row-major) * W[Cc x K](bf16,row-major)^T + bias
// One wave per 16x64 macro tile (4 accumulators share one A fragment).
// Fragment layouts per CDNA5 ISA 7.12.2:
//   A  : lane<16 row=l15 holds K {kk..kk+7, kk+16..kk+23}; lane>=16 +8/+24
//   B  : lane n holds col (n&15), same K split as A (kk+8*(n>=16), +16)
//   C/D: elem j -> row M = j + 8*(lane>=16), col N = lane&15
// ---------------------------------------------------------------------------
__global__ void __launch_bounds__(32)
bsgru_wmma_gemm_bias(const unsigned short* __restrict__ A,
                     const unsigned short* __restrict__ W,
                     const float* __restrict__ bias,
                     float* __restrict__ C, int K, int Cc) {
  const int lane = threadIdx.x & 31;
  const int l15 = lane & 15, lh = lane >> 4;
  const int tn = blockIdx.x, tm = blockIdx.y;
  const unsigned short* arow = A + (size_t)(tm * 16 + l15) * (size_t)K;
  const unsigned short* brow0 = W + (size_t)(tn * 64 + l15) * (size_t)K;
  v8f acc[4];
#pragma unroll
  for (int j = 0; j < 4; ++j)
    acc[j] = (v8f){0.f, 0.f, 0.f, 0.f, 0.f, 0.f, 0.f, 0.f};
  for (int kk = 0; kk < K; kk += 32) {
    v8bf alo = *(const v8bf*)(arow + kk + (lh << 3));
    v8bf ahi = *(const v8bf*)(arow + kk + (lh << 3) + 16);
    v16bf av = __builtin_shufflevector(alo, ahi, 0, 1, 2, 3, 4, 5, 6, 7,
                                       8, 9, 10, 11, 12, 13, 14, 15);
#pragma unroll
    for (int j = 0; j < 4; ++j) {
      const unsigned short* brow = brow0 + (size_t)(j * 16) * (size_t)K;
      v8bf blo = *(const v8bf*)(brow + kk + (lh << 3));
      v8bf bhi = *(const v8bf*)(brow + kk + (lh << 3) + 16);
      v16bf bv = __builtin_shufflevector(blo, bhi, 0, 1, 2, 3, 4, 5, 6, 7,
                                         8, 9, 10, 11, 12, 13, 14, 15);
      acc[j] = __builtin_amdgcn_wmma_f32_16x16x32_bf16(
          false, av, false, bv, (short)0, acc[j], false, false);
    }
  }
#pragma unroll
  for (int j = 0; j < 4; ++j) {
    const int col = tn * 64 + j * 16 + l15;
    const float bb = bias[col];
#pragma unroll
    for (int q = 0; q < 8; ++q) {
      const int row = tm * 16 + (lh << 3) + q;
      C[(size_t)row * (size_t)Cc + col] = acc[j][q] + bb;
    }
  }
}

// ---------------------------------------------------------------------------
// kx[r,m] = X[r,:] . Wk[m,:] + bk[m]   (tiny: R*M outputs)
// ---------------------------------------------------------------------------
__global__ void bsgru_kx(const float* __restrict__ X,
                         const float* __restrict__ Wk,
                         const float* __restrict__ bk,
                         float* __restrict__ out, int K) {
  const int gid = blockIdx.x * blockDim.x + threadIdx.x;
  if (gid >= R_ * M_) return;
  const int r = gid >> 3, m = gid & 7;
  const float* xr = X + (size_t)r * K;
  const float* wr = Wk + (size_t)m * K;
  float s = bk[m];
  for (int k = 0; k < K; ++k) s += xr[k] * wr[k];
  out[gid] = s;
}

// ---------------------------------------------------------------------------
// Device-scope barrier (generation counter). count must be 0 at kernel entry;
// every completed barrier leaves it 0 again.
// ---------------------------------------------------------------------------
__device__ __forceinline__ void grid_sync_(unsigned* count, unsigned* gen,
                                           unsigned nb) {
  __threadfence();
  __syncthreads();
  if (threadIdx.x == 0) {
    unsigned g = __hip_atomic_load(gen, __ATOMIC_RELAXED,
                                   __HIP_MEMORY_SCOPE_AGENT);
    unsigned arrived = __hip_atomic_fetch_add(count, 1u, __ATOMIC_ACQ_REL,
                                              __HIP_MEMORY_SCOPE_AGENT);
    if (arrived == nb - 1u) {
      __hip_atomic_store(count, 0u, __ATOMIC_RELAXED,
                         __HIP_MEMORY_SCOPE_AGENT);
      __hip_atomic_fetch_add(gen, 1u, __ATOMIC_RELEASE,
                             __HIP_MEMORY_SCOPE_AGENT);
    } else {
      while (__hip_atomic_load(gen, __ATOMIC_ACQUIRE,
                               __HIP_MEMORY_SCOPE_AGENT) == g)
        __builtin_amdgcn_s_sleep(2);
    }
  }
  __syncthreads();
  __threadfence();
}

// ---------------------------------------------------------------------------
// Persistent scan kernel: GSC workgroups x 256 threads (8 waves).
// WG g owns h columns [g*HS_, g*HS_+HS_). LDS holds the 96 needed W_hh rows
// (z/r/n x HS_) in bf16 (192KB, staged via async-to-LDS) + one W_hk row.
// Waves 0..5: six 16x16 WMMA tiles of gh over K=H. Waves 6..7: k-gate dots.
// ---------------------------------------------------------------------------
__global__ void __launch_bounds__(256)
bsgru_scan(const float* __restrict__ gx,             // (R_, 3H)
           const float* __restrict__ kx,             // (R_, M)
           const unsigned short* __restrict__ whhb,  // (3H, H) bf16
           const float* __restrict__ whk,            // (M, H)
           const float* __restrict__ bhh,            // (3H)
           const float* __restrict__ bhk,            // (M)
           const unsigned short* __restrict__ zbuf,  // 64KB zeros
           float* __restrict__ hout,                 // (B,T,H) f32
           unsigned short* __restrict__ hbf,         // (B,T,H) bf16
           unsigned* bar_count, unsigned* bar_gen) {
  extern __shared__ char smem[];
  unsigned short* wsl = (unsigned short*)smem;                    // 96*H bf16
  float* whk_l = (float*)(smem + (size_t)96 * H_ * 2);            // H floats
  float* ghT   = whk_l + H_;                                      // 6*256
  float* kpart = ghT + 6 * 256;                                   // 64
  float* kval  = kpart + 64;                                      // 16

  const int tid = threadIdx.x;
  const int g = blockIdx.x;
  const int wave = tid >> 5, lane = tid & 31;
  const int l15 = lane & 15, lh = lane >> 4;
  const int mblk = (g * HS_) / LB_;

  // ---- stage the resident W_hh slice into LDS with async b128 copies ----
  {
    const unsigned wsl_base = lds_off_((void*)wsl);
    const char* gbase = (const char*)whhb;
    // 96 rows x 2048 B = 12288 x 16B transfers, 48 per thread
    for (int i = tid; i < 12288; i += 256) {
      const int o = i << 4;                       // byte offset within slice
      const int rho = o >> 11, cu = o & 2047;     // row, byte-in-row
      const int gate = rho >> 5, c = rho & 31;
      const size_t gofs =
          ((size_t)(gate * H_ + g * HS_ + c) << 11) + (size_t)cu;
      const unsigned ldsaddr = wsl_base + (unsigned)o;
      const unsigned long long ga = (unsigned long long)(gbase + gofs);
      asm volatile("global_load_async_to_lds_b128 %0, %1, off"
                   :: "v"(ldsaddr), "v"(ga)
                   : "memory");
    }
    for (int i = tid; i < H_; i += 256)
      whk_l[i] = whk[(size_t)mblk * H_ + i];
    asm volatile("s_wait_asynccnt 0x0" ::: "memory");
  }
  __syncthreads();

  const float* zbuf_f = (const float*)zbuf;

  for (int t = 0; t < T_; ++t) {
    const unsigned short* hb;
    const float* hf;
    size_t hsb, hsf;
    if (t == 0) {
      hb = zbuf;   hsb = H_;
      hf = zbuf_f; hsf = H_;
    } else {
      hb = hbf + (size_t)(t - 1) * H_;  hsb = (size_t)T_ * H_;
      hf = hout + (size_t)(t - 1) * H_; hsf = (size_t)T_ * H_;
    }

    if (wave < 6) {
      // gh tile: gate = wave/2, column half = wave&1
      const int gate = wave >> 1, half = wave & 1;
      const unsigned short* arow = hb + (size_t)l15 * hsb;
      const unsigned short* brow =
          wsl + (size_t)(gate * 32 + half * 16 + l15) * H_;
      v8f acc = {0.f, 0.f, 0.f, 0.f, 0.f, 0.f, 0.f, 0.f};
      for (int kk = 0; kk < H_; kk += 32) {
        v8bf alo = *(const v8bf*)(arow + kk + (lh << 3));
        v8bf ahi = *(const v8bf*)(arow + kk + (lh << 3) + 16);
        v8bf blo = *(const v8bf*)(brow + kk + (lh << 3));        // ds_load
        v8bf bhi = *(const v8bf*)(brow + kk + (lh << 3) + 16);
        v16bf av = __builtin_shufflevector(alo, ahi, 0, 1, 2, 3, 4, 5, 6, 7,
                                           8, 9, 10, 11, 12, 13, 14, 15);
        v16bf bv = __builtin_shufflevector(blo, bhi, 0, 1, 2, 3, 4, 5, 6, 7,
                                           8, 9, 10, 11, 12, 13, 14, 15);
        acc = __builtin_amdgcn_wmma_f32_16x16x32_bf16(
            false, av, false, bv, (short)0, acc, false, false);
      }
#pragma unroll
      for (int j = 0; j < 8; ++j)
        ghT[wave * 256 + ((lh << 3) + j) * 16 + l15] = acc[j];
    } else {
      // beta-gate partial dot: h[b,:] . w_hk[mblk,:]
      const int i = tid - 192;  // 0..63
      const int b = i & 15, q = i >> 4;
      const float4* hv = (const float4*)(hf + (size_t)b * hsf + q * 256);
      const float4* wv = (const float4*)(whk_l + q * 256);
      float s = 0.f;
      for (int j = 0; j < 64; ++j) {
        float4 a = hv[j], w = wv[j];
        s += a.x * w.x + a.y * w.y + a.z * w.z + a.w * w.w;
      }
      kpart[q * 16 + b] = s;
    }
    __syncthreads();

    if (tid < 16) {
      float s = kpart[tid] + kpart[16 + tid] + kpart[32 + tid] + kpart[48 + tid];
      float kxv = kx[((size_t)tid * T_ + t) * M_ + mblk];
      kval[tid] = sigmoidf_(BETA_ * (kxv + s + bhk[mblk]));
    }
    __syncthreads();

    // gate combine + state update for this WG's 16x32 slice
    for (int e = tid; e < B_ * HS_; e += 256) {
      const int b = e >> 5, c = e & 31;
      const int half = c >> 4, cl = c & 15;
      const float ghz = ghT[(0 + half) * 256 + b * 16 + cl];
      const float ghr = ghT[(2 + half) * 256 + b * 16 + cl];
      const float ghn = ghT[(4 + half) * 256 + b * 16 + cl];
      const int col = g * HS_ + c;
      const float* gxr = gx + ((size_t)b * T_ + t) * G3_;
      if (t + 1 < T_) __builtin_prefetch(gxr + G3_ + col, 0, 1);
      const float z = sigmoidf_(gxr[col] + ghz + bhh[col]);
      const float r = sigmoidf_(gxr[H_ + col] + ghr + bhh[H_ + col]);
      const float n = tanhf(gxr[2 * H_ + col] + r * (ghn + bhh[2 * H_ + col]));
      const float hp = hf[(size_t)b * hsf + col];
      const float hn = hp + kval[b] * (1.f - z) * (n - hp);
      const size_t o = (size_t)b * T_ * H_ + (size_t)t * H_ + col;
      hout[o] = hn;
      hbf[o] = f2bf(hn);
    }
    grid_sync_(bar_count, bar_gen, GSC);
  }
}

// ---------------------------------------------------------------------------
extern "C" void kernel_launch(void* const* d_in, const int* in_sizes, int n_in,
                              void* d_out, int out_size, void* d_ws,
                              size_t ws_size, hipStream_t stream) {
  (void)in_sizes; (void)n_in; (void)out_size; (void)ws_size;
  const float* x    = (const float*)d_in[0];
  const float* wih0 = (const float*)d_in[1];
  const float* whh0 = (const float*)d_in[2];
  const float* bih0 = (const float*)d_in[3];
  const float* bhh0 = (const float*)d_in[4];
  const float* wik0 = (const float*)d_in[5];
  const float* whk0 = (const float*)d_in[6];
  const float* bik0 = (const float*)d_in[7];
  const float* bhk0 = (const float*)d_in[8];
  const float* wih1 = (const float*)d_in[9];
  const float* whh1 = (const float*)d_in[10];
  const float* bih1 = (const float*)d_in[11];
  const float* bhh1 = (const float*)d_in[12];
  const float* wik1 = (const float*)d_in[13];
  const float* whk1 = (const float*)d_in[14];
  const float* bik1 = (const float*)d_in[15];
  const float* bhk1 = (const float*)d_in[16];

  char* ws = (char*)d_ws;
  size_t off = 0;
  auto alloc = [&](size_t bytes) {
    size_t o = off;
    off += (bytes + 255) & ~(size_t)255;
    return o;
  };
  const size_t o_zero = alloc(65536);                    // h0 zeros (bf16+f32)
  const size_t o_bar  = alloc(256);                      // barrier count, gen
  const size_t o_xbf  = alloc((size_t)R_ * N_ * 2);
  const size_t o_wi0  = alloc((size_t)G3_ * N_ * 2);
  const size_t o_wh0  = alloc((size_t)G3_ * H_ * 2);
  const size_t o_wi1  = alloc((size_t)G3_ * H_ * 2);
  const size_t o_wh1  = alloc((size_t)G3_ * H_ * 2);
  const size_t o_gx   = alloc((size_t)R_ * G3_ * 4);     // 96 MiB, reused
  const size_t o_kx0  = alloc((size_t)R_ * M_ * 4);
  const size_t o_kx1  = alloc((size_t)R_ * M_ * 4);
  const size_t o_h1f  = alloc((size_t)R_ * H_ * 4);
  const size_t o_h1b  = alloc((size_t)R_ * H_ * 2);
  const size_t o_h2b  = alloc((size_t)R_ * H_ * 2);      // total ~185 MiB

  unsigned* zerop = (unsigned*)(ws + o_zero);
  unsigned* barc  = (unsigned*)(ws + o_bar);
  unsigned* barg  = barc + 1;
  unsigned short* xbf  = (unsigned short*)(ws + o_xbf);
  unsigned short* wi0b = (unsigned short*)(ws + o_wi0);
  unsigned short* wh0b = (unsigned short*)(ws + o_wh0);
  unsigned short* wi1b = (unsigned short*)(ws + o_wi1);
  unsigned short* wh1b = (unsigned short*)(ws + o_wh1);
  float* gxp  = (float*)(ws + o_gx);
  float* kx0p = (float*)(ws + o_kx0);
  float* kx1p = (float*)(ws + o_kx1);
  float* h1f  = (float*)(ws + o_h1f);
  unsigned short* h1b = (unsigned short*)(ws + o_h1b);
  unsigned short* h2b = (unsigned short*)(ws + o_h2b);

  const size_t smem = (size_t)96 * H_ * 2 + H_ * 4 + 6 * 256 * 4 + 64 * 4 + 64;
  hipFuncSetAttribute((const void*)bsgru_scan,
                      hipFuncAttributeMaxDynamicSharedMemorySize, (int)smem);

  // init: zeros page + barrier count (gen is free-running)
  bsgru_zero_u32<<<64, 256, 0, stream>>>(zerop, 65536 / 4);
  bsgru_zero_u32<<<1, 32, 0, stream>>>(barc, 1);

  // bf16 conversions
  bsgru_cvt_bf16<<<2048, 256, 0, stream>>>(x, xbf, R_ * N_);
  bsgru_cvt_bf16<<<2048, 256, 0, stream>>>(wih0, wi0b, G3_ * N_);
  bsgru_cvt_bf16<<<2048, 256, 0, stream>>>(whh0, wh0b, G3_ * H_);
  bsgru_cvt_bf16<<<2048, 256, 0, stream>>>(wih1, wi1b, G3_ * H_);
  bsgru_cvt_bf16<<<2048, 256, 0, stream>>>(whh1, wh1b, G3_ * H_);

  const dim3 gg(G3_ / 64, R_ / 16);

  // ---- layer 0 ----
  bsgru_wmma_gemm_bias<<<gg, 32, 0, stream>>>(xbf, wi0b, bih0, gxp, N_, G3_);
  bsgru_kx<<<(R_ * M_ + 255) / 256, 256, 0, stream>>>(x, wik0, bik0, kx0p, N_);
  bsgru_scan<<<GSC, 256, smem, stream>>>(gxp, kx0p, wh0b, whk0, bhh0, bhk0,
                                         (const unsigned short*)zerop, h1f,
                                         h1b, barc, barg);

  // ---- layer 1 ----
  bsgru_wmma_gemm_bias<<<gg, 32, 0, stream>>>(h1b, wi1b, bih1, gxp, H_, G3_);
  bsgru_kx<<<(R_ * M_ + 255) / 256, 256, 0, stream>>>(h1f, wik1, bik1, kx1p, H_);
  bsgru_scan<<<GSC, 256, smem, stream>>>(gxp, kx1p, wh1b, whk1, bhh1, bhk1,
                                         (const unsigned short*)zerop,
                                         (float*)d_out, h2b, barc, barg);
}
